// GNN_44281112822299
// MI455X (gfx1250) — matrix operand order repaired and checked by reading, not compiled, not run on hardware
//
#include <hip/hip_runtime.h>
#include <hip/hip_bf16.h>

typedef __attribute__((ext_vector_type(16))) __bf16 v16bf;
typedef __attribute__((ext_vector_type(8)))  __bf16 v8bf;
typedef __attribute__((ext_vector_type(4)))  __bf16 v4bf;
typedef __attribute__((ext_vector_type(8)))  float  v8f;

#define NFEAT0 192
#define H1DIM  128
#define H2DIM  64

// ---------------- degree / norm ----------------

__global__ void deg_init_kernel(float* deg, int n) {
    int i = blockIdx.x * blockDim.x + threadIdx.x;
    if (i < n) deg[i] = 1.0f;               // self loop
}

__global__ void deg_count_kernel(const int* __restrict__ dst, float* deg, int E) {
    int e = blockIdx.x * blockDim.x + threadIdx.x;
    if (e < E) atomicAdd(&deg[dst[e]], 1.0f);
}

__global__ void dinv_kernel(float* deg, int n) {
    int i = blockIdx.x * blockDim.x + threadIdx.x;
    if (i < n) deg[i] = rsqrtf(deg[i]);     // deg >= 1 always
}

// ---------------- weight pre-pack into per-lane WMMA B-fragment order ----------
// Wp layout: [(ct*(K/32) + kk)*32 + lane][16 bf16] -> one 32B load per lane.

__global__ void pack_w_kernel(const float* __restrict__ W, __bf16* __restrict__ Wp,
                              int K, int ncols)
{
    int idx = blockIdx.x * blockDim.x + threadIdx.x;
    if (idx >= K * ncols) return;
    int j      = idx & 15;
    int lane   = (idx >> 4) & 31;
    int ksteps = K >> 5;
    int kk     = (idx >> 9) % ksteps;
    int ct     = idx / (ksteps << 9);
    int krow   = kk * 32 + ((lane >> 4) << 3) + (j & 7) + ((j >> 3) << 4);  // B: K = 8h+j (j<8), 16+8h+(j-8) (j>=8)
    int n      = (lane & 15) + ct * 16;             // B: column = lane&15 (+tile)
    Wp[idx] = (__bf16)W[(size_t)krow * ncols + n];
}

// ---------------- GEMM1: gather(192) @ W1 -> xw1 [N,128] and acc = xw1*dinv^2 ----

__global__ __launch_bounds__(256) void gemm1_kernel(
    const int* __restrict__ cat_ids, const int* __restrict__ sub_ids,
    const int* __restrict__ elem_ids,
    const float* __restrict__ cat_tab, const float* __restrict__ sub_tab,
    const float* __restrict__ elem_tab,
    const __bf16* __restrict__ Wp1, const float* __restrict__ dinv,
    float* __restrict__ xw1, float* __restrict__ acc, int nnodes)
{
    __shared__ __align__(16) __bf16 A[16][NFEAT0 + 8];   // stride 400B (16B multiple)
    __shared__ float Dv[16];
    const int row0 = blockIdx.x * 16;

    if (threadIdx.x < 16) {
        int node = row0 + threadIdx.x;
        Dv[threadIdx.x] = (node < nnodes) ? dinv[node] : 0.f;
    }
    // fused embedding gather: 16 rows x 192 feats as float4, packed bf16 into LDS
    for (int q = threadIdx.x; q < 16 * (NFEAT0 / 4); q += 256) {
        int r = q / (NFEAT0 / 4), fc = q - r * (NFEAT0 / 4);
        int f = fc * 4;
        int node = row0 + r;
        float4 v = make_float4(0.f, 0.f, 0.f, 0.f);
        if (node < nnodes) {
            const float* sp;
            if (f < 64)       sp = cat_tab  + (size_t)cat_ids [node] * 64 + f;
            else if (f < 128) sp = sub_tab  + (size_t)sub_ids [node] * 64 + (f - 64);
            else              sp = elem_tab + (size_t)elem_ids[node] * 64 + (f - 128);
            v = *(const float4*)sp;
        }
        v4bf p;
        p[0] = (__bf16)v.x; p[1] = (__bf16)v.y; p[2] = (__bf16)v.z; p[3] = (__bf16)v.w;
        *(v4bf*)&A[r][f] = p;
    }
    __syncthreads();

    const int wave = threadIdx.x >> 5;
    const int lane = threadIdx.x & 31;
    const int arow    = lane & 15;          // A: M row
    const int kb_lane = (lane >> 4) * 8;    // A: K lo-half / hi-half split
    const int ncol    = (lane & 15) + wave * 16;  // D column
    const __bf16* wtile = Wp1 + (size_t)wave * (NFEAT0 / 32) * 32 * 16;

    v8f c = {};
    #pragma unroll
    for (int kk = 0; kk < NFEAT0 / 32; ++kk) {
        int kb = kk * 32 + kb_lane;
        v8bf lo = *(const v8bf*)&A[arow][kb];      // 2x ds_load_b128
        v8bf hi = *(const v8bf*)&A[arow][kb + 16];
        v16bf a;
        #pragma unroll
        for (int i = 0; i < 8; ++i) { a[i] = lo[i]; a[i + 8] = hi[i]; }
        v16bf b = *(const v16bf*)(wtile + ((kk * 32 + lane) << 4)); // 2x global_load_b128
        c = __builtin_amdgcn_wmma_f32_16x16x32_bf16(false, a, false, b,
                                                    (short)0, c, false, false);
    }

    // D layout: lanes 0-15 -> M=r, lanes 16-31 -> M=8+r; N = lane&15 (+tile)
    const int mbase = (lane >> 4) * 8;
    if (row0 + 16 <= nnodes) {               // fast path: whole tile in range
        #pragma unroll
        for (int r = 0; r < 8; ++r) {
            size_t o = (size_t)(row0 + mbase + r) * H1DIM + ncol;
            float cv = c[r], s = Dv[mbase + r];
            xw1[o] = cv;
            acc[o] = cv * s * s;             // fused self-loop term
        }
    } else {
        #pragma unroll
        for (int r = 0; r < 8; ++r) {
            int m = row0 + mbase + r;
            if (m < nnodes) {
                size_t o = (size_t)m * H1DIM + ncol;
                float cv = c[r], s = Dv[mbase + r];
                xw1[o] = cv;
                acc[o] = cv * s * s;
            }
        }
    }
}

// ---------------- GEMM2: h1[N,128] @ W2[128,64] -> xw2 [N,64] and acc = xw2*dinv^2 --

__global__ __launch_bounds__(128) void gemm2_kernel(
    const float* __restrict__ h1, const __bf16* __restrict__ Wp2,
    const float* __restrict__ dinv,
    float* __restrict__ xw2, float* __restrict__ acc, int nnodes)
{
    __shared__ __align__(16) __bf16 A[16][H1DIM + 8];    // stride 272B (16B multiple)
    __shared__ float Dv[16];
    const int row0 = blockIdx.x * 16;

    if (threadIdx.x < 16) {
        int node = row0 + threadIdx.x;
        Dv[threadIdx.x] = (node < nnodes) ? dinv[node] : 0.f;
    }
    for (int q = threadIdx.x; q < 16 * (H1DIM / 4); q += 128) {
        int r = q >> 5, fc = q & 31;
        int f = fc * 4;
        int node = row0 + r;
        float4 v = make_float4(0.f, 0.f, 0.f, 0.f);
        if (node < nnodes) v = *(const float4*)(h1 + (size_t)node * H1DIM + f);
        v4bf p;
        p[0] = (__bf16)v.x; p[1] = (__bf16)v.y; p[2] = (__bf16)v.z; p[3] = (__bf16)v.w;
        *(v4bf*)&A[r][f] = p;
    }
    __syncthreads();

    const int wave = threadIdx.x >> 5;
    const int lane = threadIdx.x & 31;
    const int arow    = lane & 15;
    const int kb_lane = (lane >> 4) * 8;
    const int ncol    = (lane & 15) + wave * 16;
    const __bf16* wtile = Wp2 + (size_t)wave * (H1DIM / 32) * 32 * 16;

    v8f c = {};
    #pragma unroll
    for (int kk = 0; kk < H1DIM / 32; ++kk) {
        int kb = kk * 32 + kb_lane;
        v8bf lo = *(const v8bf*)&A[arow][kb];
        v8bf hi = *(const v8bf*)&A[arow][kb + 16];
        v16bf a;
        #pragma unroll
        for (int i = 0; i < 8; ++i) { a[i] = lo[i]; a[i + 8] = hi[i]; }
        v16bf b = *(const v16bf*)(wtile + ((kk * 32 + lane) << 4));
        c = __builtin_amdgcn_wmma_f32_16x16x32_bf16(false, a, false, b,
                                                    (short)0, c, false, false);
    }

    const int mbase = (lane >> 4) * 8;
    if (row0 + 16 <= nnodes) {
        #pragma unroll
        for (int r = 0; r < 8; ++r) {
            size_t o = (size_t)(row0 + mbase + r) * H2DIM + ncol;
            float cv = c[r], s = Dv[mbase + r];
            xw2[o] = cv;
            acc[o] = cv * s * s;
        }
    } else {
        #pragma unroll
        for (int r = 0; r < 8; ++r) {
            int m = row0 + mbase + r;
            if (m < nnodes) {
                size_t o = (size_t)m * H2DIM + ncol;
                float cv = c[r], s = Dv[mbase + r];
                xw2[o] = cv;
                acc[o] = cv * s * s;
            }
        }
    }
}

// ---------------- aggregation ----------------

// one wave per edge, 128 floats: lane handles a float4 -> 4 L2-resident f32 atomics
__global__ void edge_scatter128_kernel(const int* __restrict__ src,
                                       const int* __restrict__ dst,
                                       const float* __restrict__ dinv,
                                       const float* __restrict__ xw,
                                       float* __restrict__ acc, int E)
{
    int tid  = blockIdx.x * blockDim.x + threadIdx.x;
    int e    = tid >> 5;
    int lane = tid & 31;
    if (e >= E) return;
    int s = src[e], d = dst[e];
    float w = dinv[s] * dinv[d];
    float4 v = ((const float4*)(xw + (size_t)s * H1DIM))[lane];
    float* o = acc + (size_t)d * H1DIM + lane * 4;
    atomicAdd(o + 0, v.x * w);
    atomicAdd(o + 1, v.y * w);
    atomicAdd(o + 2, v.z * w);
    atomicAdd(o + 3, v.w * w);
}

// one wave per edge, 64 floats: lane handles a float2
__global__ void edge_scatter64_kernel(const int* __restrict__ src,
                                      const int* __restrict__ dst,
                                      const float* __restrict__ dinv,
                                      const float* __restrict__ xw,
                                      float* __restrict__ acc, int E)
{
    int tid  = blockIdx.x * blockDim.x + threadIdx.x;
    int e    = tid >> 5;
    int lane = tid & 31;
    if (e >= E) return;
    int s = src[e], d = dst[e];
    float w = dinv[s] * dinv[d];
    float2 v = ((const float2*)(xw + (size_t)s * H2DIM))[lane];
    float* o = acc + (size_t)d * H2DIM + lane * 2;
    atomicAdd(o + 0, v.x * w);
    atomicAdd(o + 1, v.y * w);
}

__global__ void bias_relu_kernel(float* __restrict__ h, const float* __restrict__ b,
                                 int total, int F)
{
    int idx = blockIdx.x * blockDim.x + threadIdx.x;
    if (idx >= total) return;
    float v = h[idx] + b[idx & (F - 1)];     // F is a power of two
    h[idx] = v > 0.0f ? v : 0.0f;
}

// ---------------- launcher ----------------

extern "C" void kernel_launch(void* const* d_in, const int* in_sizes, int n_in,
                              void* d_out, int out_size, void* d_ws, size_t ws_size,
                              hipStream_t stream)
{
    const int*   cat_ids  = (const int*)  d_in[0];
    const int*   sub_ids  = (const int*)  d_in[1];
    const int*   elem_ids = (const int*)  d_in[2];
    const int*   edges    = (const int*)  d_in[3];
    const float* cat_tab  = (const float*)d_in[4];
    const float* sub_tab  = (const float*)d_in[5];
    const float* elem_tab = (const float*)d_in[6];
    const float* W1       = (const float*)d_in[7];
    const float* b1       = (const float*)d_in[8];
    const float* W2       = (const float*)d_in[9];
    const float* b2       = (const float*)d_in[10];
    float*       out      = (float*)d_out;

    const int N = in_sizes[0];
    const int E = in_sizes[3] / 2;
    const int* src = edges;
    const int* dst = edges + E;

    // workspace layout (64B-aligned slices)
    char*  base = (char*)d_ws;
    size_t off  = 0;
    auto carve = [&](size_t bytes) {
        void* p = base + off;
        off = (off + bytes + 63) & ~(size_t)63;
        return p;
    };
    float*  buf0 = (float*) carve((size_t)N * H1DIM * sizeof(float)); // xw1, then xw2
    float*  buf1 = (float*) carve((size_t)N * H1DIM * sizeof(float)); // h1 accumulator
    float*  dinv = (float*) carve((size_t)N * sizeof(float));         // deg -> dinv
    __bf16* Wp1  = (__bf16*)carve((size_t)NFEAT0 * H1DIM * sizeof(__bf16));
    __bf16* Wp2  = (__bf16*)carve((size_t)H1DIM  * H2DIM * sizeof(__bf16));

    const int T = 256;
    auto blocks = [](long work, int t) { return (unsigned)((work + t - 1) / t); };

    // degrees -> dinv (in place) + weight pre-pack (independent, small)
    deg_init_kernel <<<blocks(N, T), T, 0, stream>>>(dinv, N);
    deg_count_kernel<<<blocks(E, T), T, 0, stream>>>(dst, dinv, E);
    dinv_kernel     <<<blocks(N, T), T, 0, stream>>>(dinv, N);
    pack_w_kernel   <<<blocks((long)NFEAT0 * H1DIM, T), T, 0, stream>>>(W1, Wp1, NFEAT0, H1DIM);
    pack_w_kernel   <<<blocks((long)H1DIM  * H2DIM, T), T, 0, stream>>>(W2, Wp2, H1DIM,  H2DIM);

    const unsigned mtiles = (unsigned)((N + 15) / 16);

    // layer 1: GEMM (+fused self-loop init of accumulator), edge scatter, bias+relu
    gemm1_kernel<<<mtiles, 256, 0, stream>>>(cat_ids, sub_ids, elem_ids,
                                             cat_tab, sub_tab, elem_tab,
                                             Wp1, dinv, buf0, buf1, N);
    edge_scatter128_kernel<<<blocks((long)E * 32, T), T, 0, stream>>>(
        src, dst, dinv, buf0, buf1, E);
    bias_relu_kernel<<<blocks((long)N * H1DIM, T), T, 0, stream>>>(
        buf1, b1, N * H1DIM, H1DIM);

    // layer 2 (xw2 reuses buf0; accumulator is d_out)
    gemm2_kernel<<<mtiles, 128, 0, stream>>>(buf1, Wp2, dinv, buf0, out, N);
    edge_scatter64_kernel<<<blocks((long)E * 32, T), T, 0, stream>>>(
        src, dst, dinv, buf0, out, E);
    bias_relu_kernel<<<blocks((long)N * H2DIM, T), T, 0, stream>>>(
        out, b2, N * H2DIM, H2DIM);
}
